// ActorNetwork_57226144252627
// MI455X (gfx1250) — matrix operand order repaired and checked by reading, not compiled, not run on hardware
//
#include <hip/hip_runtime.h>
#include <math.h>

#define BATCH 16384

typedef __attribute__((ext_vector_type(16))) __bf16 v16bf;
typedef __attribute__((ext_vector_type(8)))  float  v8f;
typedef unsigned short u16;

union Frag16 {
  v16bf v;
  uint4 q[2];
};

__device__ __forceinline__ u16 f2bf(float f) {
  unsigned u = __float_as_uint(f);
  u += 0x7FFFu + ((u >> 16) & 1u);   // round-to-nearest-even
  return (u16)(u >> 16);
}

// ---------------------------------------------------------------------------
// GEMM: C[M,N] = act( A[M,K]_bf16 @ W[N,K]_bf16^T + bias_f32 )
// Fragments loaded as raw b128s (layouts per ISA 7.12.2); f32 accumulation in
// v_wmma_f32_16x16x32_bf16. Branch-free, fully unconditional software
// pipeline: the A fragment for k+32 and the tile-0 B fragment for k+32 are
// prefetched while the current k-step's wmmas run; on the last step the
// prefetch harmlessly wraps to k=0 (one dead load per GEMM, no branches, no
// conditional register copies). K must be a multiple of 64 here (2048 / 256 /
// 128 / 192 all are); launch covers N exactly: gridDim.y * NT * 16 == N.
// Block = 128 threads = 4 waves; wave owns a 16-row strip x NT 16-col tiles.
// ACT: 0=none, 1=relu, 2=tanh.  MASK: multiply by rowmask[row].
// OUTBF: store bf16 (else f32).
// ---------------------------------------------------------------------------
template <int ACT, bool MASK, int NT, bool OUTBF>
__global__ __launch_bounds__(128) void gemm_bf16_wmma(
    const u16* __restrict__ A, int lda,
    const u16* __restrict__ W, int ldb,
    const float* __restrict__ bias,
    void* __restrict__ Cv, int ldc,
    int K,
    const float* __restrict__ rowmask)
{
  const int wave = threadIdx.x >> 5;
  const int lane = threadIdx.x & 31;
  const int h    = lane >> 4;        // lane half (K-half select)
  const int ln   = lane & 15;        // row (A) / col (B,C) within tile
  const int m0   = (blockIdx.x * 4 + wave) * 16;
  const int n00  = blockIdx.y * (NT * 16);

  const v8f vzero = {0.f, 0.f, 0.f, 0.f, 0.f, 0.f, 0.f, 0.f};
  v8f acc[NT];
#pragma unroll
  for (int t = 0; t < NT; ++t) acc[t] = vzero;

  // A fragment (16x32 bf16): lane holds K = {k0+8h.., k0+16+8h..} (8+8)
  const u16* arow  = A + (size_t)(m0 + ln) * lda + 8 * h;
  // B fragment (32x16 bf16): lane holds col n0+ln, K = {k0+8h.., k0+16+8h..} (8+8)
  const u16* wrow0 = W + (size_t)(n00 + ln) * ldb + 8 * h;

  auto loadA = [&](int k0) {
    Frag16 f;
    f.q[0] = *(const uint4*)(arow + k0);
    f.q[1] = *(const uint4*)(arow + k0 + 16);
    return f;
  };
  auto loadB = [&](int k0, int t) {
    const u16* p = wrow0 + (size_t)t * 16 * ldb + k0;
    Frag16 f;
    f.q[0] = ((const uint4*)p)[0];
    f.q[1] = ((const uint4*)p)[2];
    return f;
  };

  Frag16 a_cur = loadA(0);
  Frag16 b_cur = loadB(0, 0);

#pragma unroll 2
  for (int k0 = 0; k0 < K; k0 += 32) {
    const int k_nxt = (k0 + 32 < K) ? (k0 + 32) : 0;  // uniform SALU select
    Frag16 a_nxt = loadA(k_nxt);                      // unconditional prefetch
#pragma unroll
    for (int t = 0; t < NT; ++t) {
      Frag16 b_nxt = (t + 1 < NT) ? loadB(k0, t + 1) : loadB(k_nxt, 0);
      acc[t] = __builtin_amdgcn_wmma_f32_16x16x32_bf16(
          false, a_cur.v, false, b_cur.v, (short)0, acc[t], false, false);
      b_cur = b_nxt;
    }
    a_cur = a_nxt;
  }

  // Epilogue. C layout: VGPR r <-> row m0 + r + 8h, col n0 + ln.
  float rm[8];
  if (MASK) {
#pragma unroll
    for (int r = 0; r < 8; ++r) rm[r] = rowmask[m0 + 8 * h + r];
  }

#pragma unroll
  for (int t = 0; t < NT; ++t) {
    int col = n00 + t * 16 + ln;
    float bc = bias[col];
#pragma unroll
    for (int r = 0; r < 8; ++r) {
      int row = m0 + 8 * h + r;
      float v = acc[t][r] + bc;
      if (ACT == 1) v = fmaxf(v, 0.0f);
      else if (ACT == 2) v = tanhf(v);
      if (MASK) v *= rm[r];
      if (OUTBF) ((u16*)Cv)[(size_t)row * ldc + col] = f2bf(v);
      else       ((float*)Cv)[(size_t)row * ldc + col] = v;
    }
  }
}

// ---------------------------------------------------------------------------
// Bulk f32 -> packed bf16x2 conversion (no padding); n2 = elements/2.
// ---------------------------------------------------------------------------
__global__ void cvt_bulk_kernel(const float* __restrict__ src,
                                unsigned* __restrict__ dst, long n2) {
  long i = blockIdx.x * (long)blockDim.x + threadIdx.x;
  if (i >= n2) return;
  float2 f = ((const float2*)src)[i];
  unsigned lo = f2bf(f.x), hi = f2bf(f.y);
  dst[i] = lo | (hi << 16);
}

// f32 -> bf16 with zero padding to [rows_dst, cols_dst]
__global__ void cvt_pad_kernel(const float* __restrict__ src,
                               u16* __restrict__ dst,
                               int rows_src, int cols_src,
                               int rows_dst, int cols_dst) {
  int i = blockIdx.x * blockDim.x + threadIdx.x;
  if (i >= rows_dst * cols_dst) return;
  int r = i / cols_dst, c = i % cols_dst;
  float v = (r < rows_src && c < cols_src) ? src[r * cols_src + c] : 0.0f;
  dst[i] = f2bf(v);
}

__global__ void pad_bias_kernel(const float* __restrict__ src, float* __restrict__ dst,
                                int n_src, int n_dst) {
  int i = blockIdx.x * blockDim.x + threadIdx.x;
  if (i < n_dst) dst[i] = (i < n_src) ? src[i] : 0.0f;
}

// ---------------------------------------------------------------------------
// Per-row masked argmax over 512 f32 logits (one wave per row).
// ---------------------------------------------------------------------------
__global__ __launch_bounds__(128) void argmax_onehot_kernel(
    const float* __restrict__ logits, const float* __restrict__ tmask,
    const int* __restrict__ ttypes, float* __restrict__ onehot,
    int* __restrict__ idx_out, float* __restrict__ rowmask)
{
  const int row  = blockIdx.x * 4 + (threadIdx.x >> 5);
  const int lane = threadIdx.x & 31;
  const float* lr = logits + (size_t)row * 512;
  const float* mr = tmask  + (size_t)row * 512;

  float bv = -INFINITY;
  int   bi = 0;
  for (int c = lane; c < 512; c += 32) {
    float v = lr[c] + (1.0f - mr[c]) * -1e9f;
    if (v > bv || (v == bv && c < bi)) { bv = v; bi = c; }
  }
  for (int off = 16; off > 0; off >>= 1) {
    float ov = __shfl_xor(bv, off, 32);
    int   oi = __shfl_xor(bi, off, 32);
    if (ov > bv || (ov == bv && oi < bi)) { bv = ov; bi = oi; }
  }

  float* oh = onehot + (size_t)row * 512;
  for (int c = lane; c < 512; c += 32) oh[c] = (c == bi) ? 1.0f : 0.0f;
  if (lane == 0) {
    idx_out[row] = bi;
    rowmask[row] = (ttypes[bi] == 1) ? 1.0f : 0.0f;
  }
}

// ---------------------------------------------------------------------------
// Z1 = bf16( relu( Z1pre + W1p_f32[c, 2048 + idx[row]] ) )   (pi layer-1
// one-hot contribution, gathered at full f32 precision)
// ---------------------------------------------------------------------------
__global__ void gather_relu_kernel(const float* __restrict__ Z1pre,
                                   const float* __restrict__ W1p,
                                   const int* __restrict__ idx,
                                   u16* __restrict__ Z1) {
  int i = blockIdx.x * blockDim.x + threadIdx.x;   // [0, B*256)
  int row = i >> 8, c = i & 255;
  float v = Z1pre[i] + W1p[(size_t)c * 2560 + 2048 + idx[row]];
  Z1[i] = f2bf(fmaxf(v, 0.0f));
}

extern "C" void kernel_launch(void* const* d_in, const int* in_sizes, int n_in,
                              void* d_out, int out_size, void* d_ws, size_t ws_size,
                              hipStream_t stream) {
  (void)in_sizes; (void)n_in; (void)out_size; (void)ws_size;

  const float* state  = (const float*)d_in[0];   // [B,2048]
  const float* tmask  = (const float*)d_in[1];   // [B,512]
  const int*   ttypes = (const int*)  d_in[2];   // [512]
  const float* W1f = (const float*)d_in[3];  const float* b1f = (const float*)d_in[4];
  const float* W2f = (const float*)d_in[5];  const float* b2f = (const float*)d_in[6];
  const float* W3f = (const float*)d_in[7];  const float* b3f = (const float*)d_in[8];
  const float* W4f = (const float*)d_in[9];  const float* b4f = (const float*)d_in[10];
  const float* W1p = (const float*)d_in[11]; const float* b1p = (const float*)d_in[12];
  const float* W2p = (const float*)d_in[13]; const float* b2p = (const float*)d_in[14];
  const float* W3p = (const float*)d_in[15]; const float* b3p = (const float*)d_in[16];
  const float* W4p = (const float*)d_in[17]; const float* b4p = (const float*)d_in[18];

  char* ws = (char*)d_ws;
  const size_t MB = 1u << 20;
  // [0,64)   state_bf (dead after pi layer-1 GEMM)
  //   [0,32)   LG f32 (written after state is dead); then Z3 bf16
  // [64,72)  Y1 bf16; then Z1 bf16 (after Y1 dead)
  // [72,80)  Y2 bf16 [72,76) + Y3 bf16 [76,80); then Z2 bf16 [72,80)
  // [80,96)  Z1pre f32
  // [96,..)  IDX, RM, bf16 weight copies, padded bias
  u16*   state_bf = (u16*)(ws + 0 * MB);
  float* LG       = (float*)(ws + 0 * MB);
  u16*   Z3       = (u16*)(ws + 0 * MB);
  u16*   Y1       = (u16*)(ws + 64 * MB);
  u16*   Z1       = (u16*)(ws + 64 * MB);
  u16*   Y2       = (u16*)(ws + 72 * MB);
  u16*   Y3       = (u16*)(ws + 76 * MB);
  u16*   Z2       = (u16*)(ws + 72 * MB);
  float* Z1pre    = (float*)(ws + 80 * MB);
  int*   IDX      = (int*)  (ws + 96 * MB);
  float* RM       = (float*)(ws + 96 * MB + 256 * 1024);
  u16*   W1f_bf   = (u16*)(ws + 97 * MB);    // [256,2048]  1MB
  u16*   W1p_bf   = (u16*)(ws + 98 * MB);    // [256,2560]  1.25MB
  u16*   W4pad_bf = (u16*)(ws + 100 * MB);   // [1024,192]  384KB
  u16*   W2f_bf   = (u16*)(ws + 101 * MB);   // [128,256]
  u16*   W3f_bf   = (u16*)(ws + 102 * MB);   // [128,128]
  u16*   W4f_bf   = (u16*)(ws + 103 * MB);   // [512,128]
  u16*   W2p_bf   = (u16*)(ws + 104 * MB);   // [256,256]
  u16*   W3pad_bf = (u16*)(ws + 105 * MB);   // [192,256]
  float* b3pad    = (float*)(ws + 106 * MB); // [192]

  float* out_onehot = (float*)d_out;                         // [B,512]
  float* out_r2     = (float*)d_out + (size_t)BATCH * 512;   // [B,1024]

  const dim3 blk(128);
  const int gx = BATCH / 64;   // 4 waves/block * 16 rows/wave

  // --- one-time f32 -> bf16 conversions (streamed once) ---
  auto cvt = [&](const float* s, u16* d, long n) {
    long n2 = n / 2;
    cvt_bulk_kernel<<<dim3((unsigned)((n2 + 255) / 256)), dim3(256), 0, stream>>>(
        s, (unsigned*)d, n2);
  };
  cvt(state, state_bf, (long)BATCH * 2048);
  cvt(W1f, W1f_bf, 256L * 2048);
  cvt(W2f, W2f_bf, 128L * 256);
  cvt(W3f, W3f_bf, 128L * 128);
  cvt(W4f, W4f_bf, 512L * 128);
  cvt(W1p, W1p_bf, 256L * 2560);
  cvt(W2p, W2p_bf, 256L * 256);
  cvt_pad_kernel<<<dim3((192 * 256 + 255) / 256), dim3(256), 0, stream>>>(
      W3p, W3pad_bf, 167, 256, 192, 256);
  cvt_pad_kernel<<<dim3((1024 * 192 + 255) / 256), dim3(256), 0, stream>>>(
      W4p, W4pad_bf, 1024, 167, 1024, 192);
  pad_bias_kernel<<<dim3(1), dim3(256), 0, stream>>>(b3p, b3pad, 167, 192);

  // --- f-net layer 1 + pi-net layer 1 (both consume state_bf) ---
  gemm_bf16_wmma<1, false, 16, true><<<dim3(gx, 1), blk, 0, stream>>>(
      state_bf, 2048, W1f_bf, 2048, b1f, Y1, 256, 2048, nullptr);
  gemm_bf16_wmma<0, false, 16, false><<<dim3(gx, 1), blk, 0, stream>>>(
      state_bf, 2048, W1p_bf, 2560, b1p, Z1pre, 256, 2048, nullptr);

  // --- f-net layers 2..4 (state_bf now dead; LG overlays it) ---
  gemm_bf16_wmma<1, false, 8, true><<<dim3(gx, 1), blk, 0, stream>>>(
      Y1, 256, W2f_bf, 256, b2f, Y2, 128, 256, nullptr);
  gemm_bf16_wmma<1, false, 8, true><<<dim3(gx, 1), blk, 0, stream>>>(
      Y2, 128, W3f_bf, 128, b3f, Y3, 128, 128, nullptr);
  gemm_bf16_wmma<0, false, 16, false><<<dim3(gx, 2), blk, 0, stream>>>(
      Y3, 128, W4f_bf, 128, b4f, LG, 512, 128, nullptr);

  // --- masked argmax -> one-hot output + template index + bimolecular mask ---
  argmax_onehot_kernel<<<dim3(BATCH / 4), blk, 0, stream>>>(
      LG, tmask, ttypes, out_onehot, IDX, RM);

  // --- pi-net: fold one-hot column gather (f32) + relu, then layers 2..4 ---
  gather_relu_kernel<<<dim3(BATCH * 256 / 256), dim3(256), 0, stream>>>(
      Z1pre, W1p, IDX, Z1);
  gemm_bf16_wmma<1, false, 16, true><<<dim3(gx, 1), blk, 0, stream>>>(
      Z1, 256, W2p_bf, 256, b2p, Z2, 256, 256, nullptr);
  gemm_bf16_wmma<1, false, 12, true><<<dim3(gx, 1), blk, 0, stream>>>(
      Z2, 256, W3pad_bf, 256, b3pad, Z3, 192, 256, nullptr);
  gemm_bf16_wmma<2, true, 16, false><<<dim3(gx, 4), blk, 0, stream>>>(
      Z3, 192, W4pad_bf, 192, b4p, out_r2, 1024, 192, RM);
}
